// RNNBase_54056458388057
// MI455X (gfx1250) — matrix operand order repaired and checked by reading, not compiled, not run on hardware
//
#include <hip/hip_runtime.h>

// ---------------------------------------------------------------------------
// 2-layer LSTM for MI455X (gfx1250, wave32, WMMA + TDM).
//
// Latency-bound recurrence (2048 dependent steps) -> persistent-grid scan.
// 16 WGs each own 16 hidden units (64 gate rows = one 16-wide N-tile per gate
// i/f/g/o). The combined [W_ih | W_hh] slice (64 x 512) lives in LDS as bf16,
// and its 16 WMMA B-fragments are hoisted into VGPRs for the whole scan.
// Each step:  g = [x_t | h_{t-1}] @ Wslice^T  (v_wmma_f32_16x16x32_bf16),
// fp32 gates, c kept in LDS, h (bf16) published via ping-pong global buffer
// (L2-resident, 32 KB) + one device-wide atomic barrier per step.
// h / bf16-x tiles are staged into LDS by the Tensor Data Mover with LDS
// padding generated by the descriptor (pad_interval=128 dw, pad_amount=4 dw).
// ---------------------------------------------------------------------------

typedef __attribute__((ext_vector_type(16))) __bf16 v16bf;
typedef __attribute__((ext_vector_type(8)))  float  v8f;
typedef __attribute__((ext_vector_type(4)))  float  f4;
typedef __attribute__((ext_vector_type(4)))  unsigned int u32x4;
typedef __attribute__((ext_vector_type(8)))  int    i32x8;
typedef __attribute__((ext_vector_type(4)))  int    i32x4;

#define DEVI __device__ __forceinline__

#if __has_builtin(__builtin_amdgcn_tensor_load_to_lds) && \
    __has_builtin(__builtin_amdgcn_s_wait_tensorcnt)
#define USE_TDM 1
#else
#define USE_TDM 0
#endif

namespace {

constexpr int kB = 64;        // batch
constexpr int kT = 2048;      // timesteps
constexpr int kK = 256;       // per-layer input dim (D == H == 256)
constexpr int kH = 256;       // hidden
constexpr int kNWG = 16;      // persistent workgroups
constexpr int kThreads = 256; // 8 waves (wave32)
constexpr int kJ = 16;        // hidden units per WG
constexpr int kRows = 64;     // 4 gates * kJ gate rows per WG

// Padded LDS strides (elements) -> +4 bank offset per row: conflict-free
// 16-lane ds_load_b128 fragment reads.
constexpr int XA_STR = kK + 8;      // 264 bf16 (x / h tiles: 512B data + 16B pad)
constexpr int W_STR  = 2 * kK + 8;  // 520 bf16 (weight rows, K=512)
constexpr int G_STR  = kRows + 4;   // 68  f32
constexpr int C_STR  = kJ + 1;      // 17  f32

constexpr size_t LDS_X    = 0;
constexpr size_t LDS_H    = LDS_X + (size_t)kB * XA_STR * 2;      // 33792
constexpr size_t LDS_W    = LDS_H + (size_t)kB * XA_STR * 2;      // 67584
constexpr size_t LDS_G    = LDS_W + (size_t)kRows * W_STR * 2;    // 134144
constexpr size_t LDS_C    = LDS_G + (size_t)kB * G_STR * 4;       // 151552
constexpr size_t LDS_BIAS = LDS_C + (size_t)kB * C_STR * 4;       // 155904
constexpr size_t LDS_TOTAL = LDS_BIAS + 256;                      // 156160 B

DEVI __bf16 to_bf16(float f) {
  union { float f; unsigned u; } a; a.f = f;
  unsigned r = (a.u + 0x7FFFu + ((a.u >> 16) & 1u)) >> 16;  // RNE
  union { unsigned short s; __bf16 b; } o; o.s = (unsigned short)r;
  return o.b;
}

// A fragment (16x32 bf16, MxK), ISA 7.12.2: lanes 0-15 K={kb..kb+7,kb+16..23},
// lanes 16-31 K={kb+8..15, kb+24..31}.
DEVI v16bf load_frag_a(const __bf16* base, int stride, int row, int kb, int hi) {
  const __bf16* p = base + row * stride + kb + hi * 8;
  union { v16bf v; f4 f[2]; } u;
  u.f[0] = *(const f4*)(p);
  u.f[1] = *(const f4*)(p + 16);
  return u.v;
}

// B fragment (32x16 bf16, KxN), N-major/K-contiguous storage:
// lanes 0-15 (n=lane) K={kb..kb+7,kb+16..23} ; lanes 16-31 K={kb+8..15,kb+24..31}.
DEVI v16bf load_frag_b(const __bf16* base, int stride, int row, int kb, int hi) {
  const __bf16* p = base + row * stride + kb + hi * 8;
  union { v16bf v; f4 f[2]; } u;
  u.f[0] = *(const f4*)(p);
  u.f[1] = *(const f4*)(p + 16);
  return u.v;
}

#if USE_TDM
// 2D TDM load: global (row_dwords per row, `rows` rows, stride_dwords between
// rows) -> LDS at lds_off, inserting 4 dwords of LDS padding every 128 dwords
// (ISA 8.3/8.4 descriptor layout). Uses the 6-arg clang-23 builtin form.
DEVI void tdm_load_2d(unsigned lds_off, const void* gptr, unsigned row_dwords,
                      unsigned rows, unsigned stride_dwords) {
  unsigned long long ga = (unsigned long long)gptr;
  u32x4 g0;
  g0[0] = 1u;                                   // count=1, user descriptor
  g0[1] = lds_off;                              // lds_addr (bytes)
  g0[2] = (unsigned)(ga & 0xffffffffu);         // global_addr[31:0]
  g0[3] = (unsigned)((ga >> 32) & 0x01ffffffu)  // global_addr[56:32]
        | (2u << 30);                           // type=2 ("image")
  i32x8 g1;
  g1[0] = (int)((2u << 16)    // data_size=4B
              | (1u << 20)    // pad_enable
              | (6u << 22)    // pad_interval: 2<<6 = 128 dwords
              | (3u << 25));  // pad_amount: 3+1 = 4 dwords (16B)
  g1[1] = (int)(row_dwords << 16);   // tensor_dim0[15:0]
  g1[2] = (int)(rows << 16);         // tensor_dim0[31:16]=0 | tensor_dim1[15:0]
  g1[3] = (int)(row_dwords << 16);   // tensor_dim1[31:16]=0 | tile_dim0
  g1[4] = (int)rows;                 // tile_dim1 | tile_dim2=0
  g1[5] = (int)stride_dwords;        // tensor_dim0_stride[31:0]
  g1[6] = 0;                         // stride hi | dim1_stride lo
  g1[7] = 0;
  i32x4 z4 = {0, 0, 0, 0};
  i32x8 z8 = {0, 0, 0, 0, 0, 0, 0, 0};
  __builtin_amdgcn_tensor_load_to_lds(g0, g1, z4, z4, z8, 0);
}
#endif

DEVI void grid_barrier(unsigned* cnt, unsigned* gen, unsigned nwg) {
  __threadfence();   // release h stores to device scope
  __syncthreads();
  if (threadIdx.x == 0) {
    unsigned g = __hip_atomic_load(gen, __ATOMIC_RELAXED, __HIP_MEMORY_SCOPE_AGENT);
    unsigned a = __hip_atomic_fetch_add(cnt, 1u, __ATOMIC_ACQ_REL, __HIP_MEMORY_SCOPE_AGENT);
    if (a == nwg - 1u) {
      __hip_atomic_store(cnt, 0u, __ATOMIC_RELAXED, __HIP_MEMORY_SCOPE_AGENT);
      __hip_atomic_fetch_add(gen, 1u, __ATOMIC_RELEASE, __HIP_MEMORY_SCOPE_AGENT);
    } else {
      while (__hip_atomic_load(gen, __ATOMIC_ACQUIRE, __HIP_MEMORY_SCOPE_AGENT) == g) {
        __builtin_amdgcn_s_sleep(2);
      }
    }
  }
  __syncthreads();
  __threadfence();   // acquire: invalidate stale WGP-cached h lines
}

// Manual stage of x_t (f32 -> bf16 convert) into LDS_X.
DEVI void stage_x_f32(char* sm, const float* __restrict__ xin, int t, int tid) {
  __bf16* xa = (__bf16*)(sm + LDS_X);
  int b = tid >> 2;
  int k0 = (tid & 3) * 64;
  __bf16* dst = xa + b * XA_STR + k0;
  const float* src = xin + ((size_t)b * kT + t) * kK + k0;
  #pragma unroll
  for (int i = 0; i < 64; i += 8) {
    f4 v0 = *(const f4*)(src + i);
    f4 v1 = *(const f4*)(src + i + 4);
    union { __bf16 h[8]; f4 v; } u;
    u.h[0] = to_bf16(v0[0]); u.h[1] = to_bf16(v0[1]);
    u.h[2] = to_bf16(v0[2]); u.h[3] = to_bf16(v0[3]);
    u.h[4] = to_bf16(v1[0]); u.h[5] = to_bf16(v1[1]);
    u.h[6] = to_bf16(v1[2]); u.h[7] = to_bf16(v1[3]);
    *(f4*)(dst + i) = u.v;
  }
}

// Manual bf16 row copy global -> LDS tile (fallback when no TDM builtin).
DEVI void stage_bf16(char* sm, size_t lds_base, const __bf16* __restrict__ src0,
                     size_t row_stride_elems, int tid) {
  __bf16* dst0 = (__bf16*)(sm + lds_base);
  int b = tid >> 2;
  int k0 = (tid & 3) * 64;
  __bf16* dst = dst0 + b * XA_STR + k0;
  const __bf16* src = src0 + (size_t)b * row_stride_elems + k0;
  #pragma unroll
  for (int i = 0; i < 8; ++i)
    *(f4*)(dst + i * 8) = *(const f4*)(src + i * 8);
}

// One-time: this WG's [W_ih | W_hh] rows (64 x 512) + bias -> LDS (bf16/f32).
DEVI void stage_w(char* sm, const float* __restrict__ Wih,
                  const float* __restrict__ Whh, const float* __restrict__ bias,
                  int wg, int tid) {
  __bf16* wl = (__bf16*)(sm + LDS_W);
  int rowid = tid >> 2;             // LDS row = q*16 + c
  int kseg  = (tid & 3) * 128;
  int q = rowid >> 4, c = rowid & 15;
  int rg = q * kH + wg * kJ + c;    // row in [4H, K] weights
  const float* src = (kseg < kK) ? (Wih + (size_t)rg * kK + kseg)
                                 : (Whh + (size_t)rg * kK + (kseg - kK));
  __bf16* dst = wl + rowid * W_STR + kseg;
  #pragma unroll 4
  for (int i = 0; i < 128; i += 8) {
    f4 v0 = *(const f4*)(src + i);
    f4 v1 = *(const f4*)(src + i + 4);
    union { __bf16 h[8]; f4 v; } u;
    u.h[0] = to_bf16(v0[0]); u.h[1] = to_bf16(v0[1]);
    u.h[2] = to_bf16(v0[2]); u.h[3] = to_bf16(v0[3]);
    u.h[4] = to_bf16(v1[0]); u.h[5] = to_bf16(v1[1]);
    u.h[6] = to_bf16(v1[2]); u.h[7] = to_bf16(v1[3]);
    *(f4*)(dst + i) = u.v;
  }
  if (tid < kRows) {
    int qq = tid >> 4, cc = tid & 15;
    ((float*)(sm + LDS_BIAS))[tid] = bias[qq * kH + wg * kJ + cc];
  }
}

template <typename Tin, typename Tout>
__global__ void __launch_bounds__(kThreads, 1)
lstm_scan(const Tin* __restrict__ xin, const float* __restrict__ Wih,
          const float* __restrict__ Whh, const float* __restrict__ bias,
          Tout* __restrict__ hseq, float* __restrict__ hlast,
          float* __restrict__ clast, __bf16* __restrict__ hping,
          unsigned* __restrict__ bar) {
  extern __shared__ char sm[];
  const int tid  = threadIdx.x;
  const int wg   = blockIdx.x;
  const int lane = tid & 31;
  const int wave = tid >> 5;
  const int q    = wave & 3;   // n-tile == gate (0:i 1:f 2:g 3:o)
  const int mp   = wave >> 2;  // batch-half (m-tiles {2mp, 2mp+1})
  const int hi   = lane >> 4;
  const int ln   = lane & 15;

  stage_w(sm, Wih, Whh, bias, wg, tid);
  // Zero the h tile once so the K=[256,512) half legitimately contributes 0
  // at t==0 (lets the k-loop stay fully unrolled with no branches).
  {
    f4 z = {0.f, 0.f, 0.f, 0.f};
    for (int i = tid; i < (kB * XA_STR) / 8; i += kThreads)
      ((f4*)(sm + LDS_H))[i] = z;
  }
  __syncthreads();

  const __bf16* wl = (const __bf16*)(sm + LDS_W);
  float* G  = (float*)(sm + LDS_G);
  float* cS = (float*)(sm + LDS_C);
  const float* bS = (const float*)(sm + LDS_BIAS);

  // Hoist all 16 weight B-fragments into VGPRs for the entire scan.
  v16bf bfr[16];
  #pragma unroll
  for (int kk = 0; kk < 16; ++kk)
    bfr[kk] = load_frag_b(wl, W_STR, q * 16 + ln, kk * 32, hi);

  for (int t = 0; t < kT; ++t) {
    const __bf16* hcur = hping + (size_t)(t & 1) * (kB * kH);
    __bf16* hnxt       = hping + (size_t)((t + 1) & 1) * (kB * kH);

    // ---- stage x_t and h_{t-1} into LDS ----------------------------------
#if USE_TDM
    if (tid < 32) {   // one wave issues the DMAs; TENSORcnt tracks them
      if (t > 0)
        tdm_load_2d((unsigned)(size_t)(sm + LDS_H), hcur, kH / 2, kB, kH / 2);
      if constexpr (sizeof(Tin) == 2)
        tdm_load_2d((unsigned)(size_t)(sm + LDS_X), xin + (size_t)t * kH,
                    kH / 2, kB, (unsigned)((size_t)kT * kH / 2));
    }
    if constexpr (sizeof(Tin) == 4) stage_x_f32(sm, (const float*)xin, t, tid);
    if (tid < 32) __builtin_amdgcn_s_wait_tensorcnt(0);
#else
    if (t > 0) stage_bf16(sm, LDS_H, hcur, kH, tid);
    if constexpr (sizeof(Tin) == 4) stage_x_f32(sm, (const float*)xin, t, tid);
    else stage_bf16(sm, LDS_X, (const __bf16*)xin + (size_t)t * kH,
                    (size_t)kT * kH, tid);
#endif
    __syncthreads();

    // ---- WMMA GEMM: G[64x64] = [x_t | h] (64x512) x Wslice^T -------------
    v8f acc0 = {0.f, 0.f, 0.f, 0.f, 0.f, 0.f, 0.f, 0.f};
    v8f acc1 = {0.f, 0.f, 0.f, 0.f, 0.f, 0.f, 0.f, 0.f};
    #pragma unroll
    for (int kk = 0; kk < 16; ++kk) {
      const __bf16* ab = (const __bf16*)(sm + (kk < 8 ? LDS_X : LDS_H));
      const int kloc = (kk & 7) * 32;
      v16bf fa0 = load_frag_a(ab, XA_STR, mp * 32 + ln, kloc, hi);
      v16bf fa1 = load_frag_a(ab, XA_STR, mp * 32 + 16 + ln, kloc, hi);
      acc0 = __builtin_amdgcn_wmma_f32_16x16x32_bf16(false, fa0, false, bfr[kk],
                                                     (short)0, acc0, false, false);
      acc1 = __builtin_amdgcn_wmma_f32_16x16x32_bf16(false, fa1, false, bfr[kk],
                                                     (short)0, acc1, false, false);
    }

    // Spill C tiles to gate buffer (lane L: n = L%16, m = vgpr + 8*(L/16)).
    {
      const int n = q * 16 + ln;
      const int mb = hi * 8;
      #pragma unroll
      for (int r = 0; r < 8; ++r) {
        G[(mp * 32 + mb + r) * G_STR + n]      = acc0[r];
        G[(mp * 32 + 16 + mb + r) * G_STR + n] = acc1[r];
      }
    }
    __syncthreads();

    // ---- fp32 gates + state update ---------------------------------------
    {
      const int j  = tid & 15;
      const int b0 = tid >> 4;
      const int jg = wg * kJ + j;
      #pragma unroll
      for (int ii = 0; ii < 4; ++ii) {
        const int b = b0 + ii * 16;
        float gi = G[b * G_STR + j]      + bS[j];
        float gf = G[b * G_STR + 16 + j] + bS[16 + j];
        float gc = G[b * G_STR + 32 + j] + bS[32 + j];
        float go = G[b * G_STR + 48 + j] + bS[48 + j];
        float ig = 1.f / (1.f + __expf(-gi));
        float fg = 1.f / (1.f + __expf(-gf));
        float cg = tanhf(gc);
        float og = 1.f / (1.f + __expf(-go));
        float cp = (t == 0) ? 0.f : cS[b * C_STR + j];
        float cn = fg * cp + ig * cg;
        float hn = og * tanhf(cn);
        cS[b * C_STR + j] = cn;
        hnxt[b * kH + jg] = to_bf16(hn);               // publish for next step
        if constexpr (sizeof(Tout) == 4) {
          hseq[((size_t)b * kT + t) * kH + jg] = hn;   // layer-2 output
        } else {
          hseq[((size_t)b * kT + t) * kH + jg] = to_bf16(hn);  // layer-1 -> ws
        }
        if (hlast != nullptr && t == kT - 1) {
          hlast[b * kH + jg] = hn;
          clast[b * kH + jg] = cn;
        }
      }
    }

    if (t + 1 < kT) grid_barrier(bar, bar + 1, kNWG);
  }
}

__global__ void init_barrier(unsigned* p) {
  if (threadIdx.x < 2) p[threadIdx.x] = 0u;
}

}  // namespace

extern "C" void kernel_launch(void* const* d_in, const int* in_sizes, int n_in,
                              void* d_out, int out_size, void* d_ws, size_t ws_size,
                              hipStream_t stream) {
  (void)in_sizes; (void)n_in; (void)out_size; (void)ws_size;

  const float* x    = (const float*)d_in[0];
  const float* Wih0 = (const float*)d_in[1];
  const float* Whh0 = (const float*)d_in[2];
  const float* b0   = (const float*)d_in[3];
  const float* Wih1 = (const float*)d_in[4];
  const float* Whh1 = (const float*)d_in[5];
  const float* b1   = (const float*)d_in[6];

  float* h2seq = (float*)d_out;                 // [B, T, H]
  float* hlast = h2seq + (size_t)kB * kT * kH;  // [B, H]
  float* clast = hlast + (size_t)kB * kH;       // [B, H]

  char* ws = (char*)d_ws;
  unsigned* bar   = (unsigned*)ws;                                   // 2 u32
  __bf16*   hping = (__bf16*)(ws + 1024);                            // 2*B*H bf16
  __bf16*   h1seq = (__bf16*)(ws + 1024 + (size_t)2 * kB * kH * 2);  // B*T*H bf16

  init_barrier<<<dim3(1), dim3(32), 0, stream>>>(bar);

  // Layer 1: f32 input -> bf16 hidden sequence in workspace.
  lstm_scan<float, __bf16><<<dim3(kNWG), dim3(kThreads), LDS_TOTAL, stream>>>(
      x, Wih0, Whh0, b0, h1seq, (float*)nullptr, (float*)nullptr, hping, bar);

  // Layer 2: bf16 input -> f32 output sequence + (h_last, c_last).
  lstm_scan<__bf16, float><<<dim3(kNWG), dim3(kThreads), LDS_TOTAL, stream>>>(
      h1seq, Wih1, Whh1, b1, h2seq, hlast, clast, hping, bar);
}
